// GPT2Attention_29953101922841
// MI455X (gfx1250) — matrix operand order repaired and checked by reading, not compiled, not run on hardware
//
#include <hip/hip_runtime.h>
#include <hip/hip_bf16.h>

typedef _Float16 half_t;
typedef __attribute__((ext_vector_type(16))) _Float16 v16h;
typedef __attribute__((ext_vector_type(8)))  _Float16 v8h;
typedef __attribute__((ext_vector_type(8)))  float    v8f;
typedef __attribute__((ext_vector_type(4)))  _Float16 h4;

#define NB   8
#define SEQ  1024
#define HID  768
#define NH   12
#define HD   64
#define MTOT (NB*SEQ)          // 8192
#define N_QKV (3*HID)          // 2304
// fold softmax scale and log2(e) into q so softmax runs in exp2 domain
#define Q_SCALE 0.18033688011112042f   // 0.125 * log2(e)

// ---------------- WMMA fragment helpers (documented gfx1250 layouts) ----------------

// A fragment (16x32 f16, M x K). Element i of lane L:
//   M = L%16, K = (i<8 ? 0 : 16) + 8*(L/16) + i%8
__device__ __forceinline__ v16h load_frag_a(const _Float16* base, int stride, int lane) {
    int m = lane & 15, g = lane >> 4;
    const _Float16* p = base + m * stride + g * 8;
    v8h lo = *(const v8h*)p;         // K = 8g + 0..7
    v8h hi = *(const v8h*)(p + 16);  // K = 16 + 8g + 0..7
    return __builtin_shufflevector(lo, hi, 0,1,2,3,4,5,6,7,8,9,10,11,12,13,14,15);
}

// B fragment (32x16 f16, K x N) read from an [N][K] (transposed) buffer.
// Element i of lane L: N = L%16, K = (i<8 ? 0 : 16) + 8*(L/16) + i%8
__device__ __forceinline__ v16h load_frag_b(const _Float16* base, int stride, int lane) {
    int n = lane & 15, g = lane >> 4;
    const _Float16* p = base + n * stride + g * 8;
    v8h lo = *(const v8h*)p;
    v8h hi = *(const v8h*)(p + 16);
    return __builtin_shufflevector(lo, hi, 0,1,2,3,4,5,6,7,8,9,10,11,12,13,14,15);
}

__device__ __forceinline__ v8f wmma_f16(v16h a, v16h b, v8f c) {
    return __builtin_amdgcn_wmma_f32_16x16x32_f16(false, a, false, b, (short)0, c, false, false);
}

__device__ __forceinline__ float redmax16(float v) {
    v = fmaxf(v, __shfl_xor(v, 1, 16));
    v = fmaxf(v, __shfl_xor(v, 2, 16));
    v = fmaxf(v, __shfl_xor(v, 4, 16));
    v = fmaxf(v, __shfl_xor(v, 8, 16));
    return v;
}
__device__ __forceinline__ float redsum16(float v) {
    v += __shfl_xor(v, 1, 16);
    v += __shfl_xor(v, 2, 16);
    v += __shfl_xor(v, 4, 16);
    v += __shfl_xor(v, 8, 16);
    return v;
}

// ---- async global -> LDS copy (CDNA5 ASYNCcnt path), 32 contiguous bytes ----
__device__ __forceinline__ void async_copy32B(unsigned lds_off, const void* g) {
    asm volatile("global_load_async_to_lds_b128 %0, %1, off\n\t"
                 "global_load_async_to_lds_b128 %0, %1, off offset:16"
                 :: "v"(lds_off), "v"(g) : "memory");
}
__device__ __forceinline__ void wait_async_le6() {
    asm volatile("s_wait_asynccnt 0x6" ::: "memory");
}
__device__ __forceinline__ void wait_async_0() {
    asm volatile("s_wait_asynccnt 0x0" ::: "memory");
}

// ---------------- Kernel 1a: f32 -> f16 convert ----------------
__global__ __launch_bounds__(256) void k_convert(const float* __restrict__ in,
                                                 half_t* __restrict__ out, int n) {
    int i = (blockIdx.x * 256 + threadIdx.x) * 4;
    if (i < n) {
        float4 f = *(const float4*)(in + i);
        h4 h; h.x = (half_t)f.x; h.y = (half_t)f.y; h.z = (half_t)f.z; h.w = (half_t)f.w;
        *(h4*)(out + i) = h;
    }
}

// ---------------- Kernel 1b: W[K][N] f32 -> Wt[N][K] f16 ----------------
__global__ __launch_bounds__(256) void k_transpose(const float* __restrict__ W,
                                                   half_t* __restrict__ Wt, int K, int N) {
    __shared__ _Float16 t[32][33];
    int tx = threadIdx.x & 31, ty = threadIdx.x >> 5;
    int k0 = blockIdx.x * 32, n0 = blockIdx.y * 32;
    #pragma unroll
    for (int i = 0; i < 32; i += 8)
        t[ty + i][tx] = (half_t)W[(size_t)(k0 + ty + i) * N + n0 + tx];
    __syncthreads();
    #pragma unroll
    for (int i = 0; i < 32; i += 8)
        Wt[(size_t)(n0 + ty + i) * K + k0 + tx] = t[tx][ty + i];
}

// ---------------- Kernels 2 & 4: tiled WMMA GEMM ----------------
// C[M x N] = A[M x K] * W + bias, W given transposed as Bt[N][K].
// Block tile 128x256, BK=32, 8 waves (2x4), each wave 64x64 (4x4 frags,
// 16 WMMAs : 16 ds_b128 per K-step). Double-buffered LDS filled by
// global_load_async_to_lds_b128, pipelined on ASYNCcnt.
// MODE 0: scatter into q (scaled, head-major), k (head-major), v transposed [d][s]
// MODE 1: f32 output row-major + bias
template <int MODE>
__global__ __launch_bounds__(256) void k_gemm(const half_t* __restrict__ A,
                                              const half_t* __restrict__ Bt,
                                              const float* __restrict__ bias,
                                              half_t* __restrict__ q_out,
                                              half_t* __restrict__ k_out,
                                              half_t* __restrict__ vt_out,
                                              float* __restrict__ out,
                                              int Ktot) {
    __shared__ __align__(16) _Float16 Abuf[2][128 * 32];   // 2 x 8 KB
    __shared__ __align__(16) _Float16 Bbuf[2][256 * 32];   // 2 x 16 KB

    const int tid  = threadIdx.x;
    const int lane = tid & 31;
    const int wid  = tid >> 5;
    const int wm   = (wid >> 2) * 64;   // 0 or 64
    const int wn   = (wid & 3) * 64;    // 0,64,128,192
    const int mblock = blockIdx.x * 128;
    const int nblock = blockIdx.y * 256;

    // staging assignment: A tile 128x32 -> thread t: row t/2, 16 halves at (t&1)*16
    //                     B tile 256x32 -> thread t: row t, 32 halves
    const int lrowA = tid >> 1;
    const int lcolA = (tid & 1) * 16;
    unsigned ldsA[2], ldsB[2];
    ldsA[0] = (unsigned)(size_t)&Abuf[0][lrowA * 32 + lcolA];
    ldsA[1] = (unsigned)(size_t)&Abuf[1][lrowA * 32 + lcolA];
    ldsB[0] = (unsigned)(size_t)&Bbuf[0][tid * 32];
    ldsB[1] = (unsigned)(size_t)&Bbuf[1][tid * 32];
    const half_t* gA = A  + (size_t)(mblock + lrowA) * Ktot + lcolA;
    const half_t* gB = Bt + (size_t)(nblock + tid)  * Ktot;

    auto issue = [&](int t) {                      // 6 async b128 per thread
        int bsel = t & 1;
        async_copy32B(ldsA[bsel],      gA + t * 32);
        async_copy32B(ldsB[bsel],      gB + t * 32);
        async_copy32B(ldsB[bsel] + 32, gB + t * 32 + 16);
    };

    v8f acc[4][4] = {};
    const int T = Ktot / 32;
    issue(0);
    for (int t = 0; t < T; ++t) {
        __syncthreads();                 // everyone done reading buf[(t+1)&1]
        if (t + 1 < T) {
            issue(t + 1);                // prefetch next tile into other buffer
            wait_async_le6();            // tile t (oldest 6) has landed
        } else {
            wait_async_0();
        }
        __syncthreads();                 // all waves' tile-t copies visible

        const _Float16* As = &Abuf[t & 1][0];
        const _Float16* Bs = &Bbuf[t & 1][0];
        v16h a[4], b[4];
        #pragma unroll
        for (int fm = 0; fm < 4; fm++)
            a[fm] = load_frag_a(As + (wm + fm * 16) * 32, 32, lane);
        #pragma unroll
        for (int fn = 0; fn < 4; fn++)
            b[fn] = load_frag_b(Bs + (wn + fn * 16) * 32, 32, lane);
        #pragma unroll
        for (int fm = 0; fm < 4; fm++)
            #pragma unroll
            for (int fn = 0; fn < 4; fn++)
                acc[fm][fn] = wmma_f16(a[fm], b[fn], acc[fm][fn]);
    }

    const int g = lane >> 4, nl = lane & 15;
    #pragma unroll
    for (int fn = 0; fn < 4; fn++) {
        int col = nblock + wn + fn * 16 + nl;
        float bv = bias[col];
        #pragma unroll
        for (int fm = 0; fm < 4; fm++) {
            #pragma unroll
            for (int r = 0; r < 8; r++) {
                int row = mblock + wm + fm * 16 + g * 8 + r;
                float v = acc[fm][fn][r] + bv;
                if (MODE == 0) {
                    int which = col / HID, hc = col % HID;
                    int head = hc >> 6, d = hc & 63;
                    int bb = row >> 10, s = row & 1023;
                    size_t hb = (size_t)(bb * NH + head);
                    if (which == 0)
                        q_out[(hb * SEQ + s) * HD + d] = (half_t)(v * Q_SCALE);
                    else if (which == 1)
                        k_out[(hb * SEQ + s) * HD + d] = (half_t)v;
                    else
                        vt_out[(hb * HD + d) * SEQ + s] = (half_t)v;
                } else {
                    out[(size_t)row * HID + col] = v;
                }
            }
        }
    }
}

// ---------------- Kernel 3: causal flash attention ----------------
// One wave per 16 query rows; exactly one diagonal (masked) chunk per tile.
template <bool MASK>
__device__ __forceinline__ void attn_chunk(int kb, int qbase, int g, int nl, int lane,
                                           const half_t* __restrict__ kp,
                                           const half_t* __restrict__ vp,
                                           v16h qa0, v16h qa1,
                                           v8f (&acc)[4], float (&mrun)[8],
                                           float (&lrun)[8], _Float16* pl) {
    v8f s[2];
    #pragma unroll
    for (int j = 0; j < 2; j++) {
        const half_t* kt0 = kp + (size_t)(kb + j * 16) * HD;
        v16h b0 = load_frag_b(kt0, HD, lane);       // dims 0..31
        v16h b1 = load_frag_b(kt0 + 32, HD, lane);  // dims 32..63
        v8f z = {};
        z = wmma_f16(qa0, b0, z);
        s[j] = wmma_f16(qa1, b1, z);
    }
    #pragma unroll
    for (int r = 0; r < 8; r++) {
        float s0 = s[0][r], s1 = s[1][r];
        if (MASK) {
            int row = qbase + g * 8 + r;
            s0 = (kb + nl      > row) ? -__builtin_inff() : s0;
            s1 = (kb + 16 + nl > row) ? -__builtin_inff() : s1;
        }
        float cm = redmax16(fmaxf(s0, s1));
        float mnew = fmaxf(mrun[r], cm);
        float alpha = exp2f(mrun[r] - mnew);
        mrun[r] = mnew;
        float p0 = exp2f(s0 - mnew);
        float p1 = exp2f(s1 - mnew);
        lrun[r] = lrun[r] * alpha + redsum16(p0 + p1);
        #pragma unroll
        for (int fn = 0; fn < 4; fn++) acc[fn][r] *= alpha;
        pl[(g * 8 + r) * 32 + nl]      = (half_t)p0;
        pl[(g * 8 + r) * 32 + 16 + nl] = (half_t)p1;
    }
    // P (C-layout) -> A-layout via per-wave LDS round trip
    v16h pa = load_frag_a(pl, 32, lane);
    #pragma unroll
    for (int fn = 0; fn < 4; fn++) {
        v16h vb = load_frag_b(vp + (size_t)(fn * 16) * SEQ + kb, SEQ, lane);
        acc[fn] = wmma_f16(pa, vb, acc[fn]);
    }
}

__global__ __launch_bounds__(256) void k_attn(const half_t* __restrict__ q,
                                              const half_t* __restrict__ k,
                                              const half_t* __restrict__ vt,
                                              half_t* __restrict__ attn_out) {
    __shared__ __align__(16) _Float16 Pl[8][16 * 32];   // per-wave P staging

    const int tid  = threadIdx.x;
    const int lane = tid & 31;
    const int wid  = tid >> 5;
    const int bh     = blockIdx.x >> 3;          // 0..95
    const int qchunk = blockIdx.x & 7;
    const int qbase  = qchunk * 128 + wid * 16;
    const int bb = bh / NH, hh = bh % NH;
    const int g = lane >> 4, nl = lane & 15;

    const half_t* qp = q  + ((size_t)bh * SEQ + qbase) * HD;
    const half_t* kp = k  + (size_t)bh * SEQ * HD;
    const half_t* vp = vt + (size_t)bh * HD * SEQ;

    v16h qa0 = load_frag_a(qp, HD, lane);        // dims 0..31
    v16h qa1 = load_frag_a(qp + 32, HD, lane);   // dims 32..63

    v8f acc[4] = {};
    float mrun[8], lrun[8];
    #pragma unroll
    for (int r = 0; r < 8; r++) { mrun[r] = -__builtin_inff(); lrun[r] = 0.0f; }

    const int nfull = qbase >> 5;        // chunks with no causal masking
    for (int c = 0; c < nfull; c++) {
        // prefetch next chunk's K (32 keys x 64 dims) and V columns, lane-spread,
        // WGP scope (locality 3) so all 8 waves sharing this (b,h) hit near cache
        int nkb = (c + 1) * 32;
        __builtin_prefetch(kp + (size_t)(nkb + lane) * HD, 0, 3);
        __builtin_prefetch(vp + (size_t)lane * SEQ + nkb, 0, 3);
        __builtin_prefetch(vp + (size_t)(lane + 32) * SEQ + nkb, 0, 3);
        attn_chunk<false>(c * 32, qbase, g, nl, lane, kp, vp, qa0, qa1,
                          acc, mrun, lrun, &Pl[wid][0]);
    }
    attn_chunk<true>(nfull * 32, qbase, g, nl, lane, kp, vp, qa0, qa1,
                     acc, mrun, lrun, &Pl[wid][0]);

    half_t* op = attn_out + (size_t)bb * SEQ * HID + (size_t)hh * HD;
    #pragma unroll
    for (int r = 0; r < 8; r++) {
        int srow = qbase + g * 8 + r;
        float inv = 1.0f / lrun[r];
        #pragma unroll
        for (int fn = 0; fn < 4; fn++)
            op[(size_t)srow * HID + fn * 16 + nl] = (half_t)(acc[fn][r] * inv);
    }
}

// ---------------- Launch ----------------
extern "C" void kernel_launch(void* const* d_in, const int* in_sizes, int n_in,
                              void* d_out, int out_size, void* d_ws, size_t ws_size,
                              hipStream_t stream) {
    const float* x     = (const float*)d_in[0];
    const float* Wqkv  = (const float*)d_in[1];
    const float* bqkv  = (const float*)d_in[2];
    const float* Wproj = (const float*)d_in[3];
    const float* bproj = (const float*)d_in[4];
    float* out = (float*)d_out;

    char* p = (char*)d_ws;
    auto carve = [&](size_t bytes) {
        void* r = (void*)p;
        p += (bytes + 255) & ~(size_t)255;
        return r;
    };
    half_t* x_h     = (half_t*)carve((size_t)MTOT * HID * 2);
    half_t* wqkv_t  = (half_t*)carve((size_t)N_QKV * HID * 2);
    half_t* wproj_t = (half_t*)carve((size_t)HID * HID * 2);
    half_t* q_h     = (half_t*)carve((size_t)NB * NH * SEQ * HD * 2);
    half_t* k_h     = (half_t*)carve((size_t)NB * NH * SEQ * HD * 2);
    half_t* vt_h    = (half_t*)carve((size_t)NB * NH * HD * SEQ * 2);
    half_t* attn_h  = (half_t*)carve((size_t)MTOT * HID * 2);

    // 1) precision conversion + weight transposes
    k_convert<<<(MTOT * HID) / 1024, 256, 0, stream>>>(x, x_h, MTOT * HID);
    k_transpose<<<dim3(HID / 32, N_QKV / 32), 256, 0, stream>>>(Wqkv, wqkv_t, HID, N_QKV);
    k_transpose<<<dim3(HID / 32, HID / 32), 256, 0, stream>>>(Wproj, wproj_t, HID, HID);

    // 2) QKV projection (async-LDS double-buffered WMMA GEMM)
    k_gemm<0><<<dim3(MTOT / 128, N_QKV / 256), 256, 0, stream>>>(
        x_h, wqkv_t, bqkv, q_h, k_h, vt_h, nullptr, HID);

    // 3) causal flash attention (WMMA)
    k_attn<<<NB * NH * (SEQ / 128), 256, 0, stream>>>(q_h, k_h, vt_h, attn_h);

    // 4) output projection (WMMA GEMM, f32 out)
    k_gemm<1><<<dim3(MTOT / 128, HID / 256), 256, 0, stream>>>(
        attn_h, wproj_t, bproj, nullptr, nullptr, nullptr, out, HID);
}
